// Transformer_11888469475434
// MI455X (gfx1250) — compile-verified, matrix operand order checked by reading, not run on hardware
//
#include <hip/hip_runtime.h>
#include <hip/hip_bf16.h>

// ---------------------------------------------------------------------------
// CDNA5 (gfx1250, wave32) Point-Transformer block.
// GEMMs on v_wmma_f32_16x16x32_bf16, double-buffered LDS tiles stored in
// fragment order so each half-fragment is a single aligned ds_load_b128.
// ---------------------------------------------------------------------------

typedef __attribute__((ext_vector_type(16))) __bf16 v16bf;
typedef __attribute__((ext_vector_type(8)))  float  v8f;

#define Bc    4
#define Nc    2048
#define Kc    16
#define CINc  128
#define DIMc  256
#define PHc   64
#define AHc   1024
#define NKc   (Nc * Kc)          // 32768 grouped columns per batch
#define CHCOL 4096               // attention chunk: 256 points * 16 neighbors
#define EPSc  1e-5f

union Frag { v16bf v; uint4 q[2]; };

// ---------------------------------------------------------------------------
// Generic WMMA GEMM: out = epilogue( W(OxI) * X(I x cols) )
//   block = 256 threads (8 waves), tile = 128 rows x 64 cols, K step 32.
//   Double-buffered LDS; A tile row-major (stride 40 -> 16B-aligned frag
//   chunks), B tile staged fragment-major [nt][k-octet][line][8].
//   Fragment element e holds K = 8*half + e (e<8) / 16 + 8*half + (e-8).
// ---------------------------------------------------------------------------
template <int XBF16>
__global__ __launch_bounds__(256)
void wmma_gemm_kernel(const float* __restrict__ W,      // O x I, row-major
                      const void*  __restrict__ X,      // I x ldx
                      const float* __restrict__ bias,
                      const float* __restrict__ bn_g, const float* __restrict__ bn_b,
                      const float* __restrict__ bn_m, const float* __restrict__ bn_v,
                      int do_relu,
                      float*  __restrict__ out_f32,
                      __bf16* __restrict__ out_bf16,
                      const float* __restrict__ resid,
                      int I, int ldx, int ldo, int ldr,
                      long long xBStride, long long oBStride, long long rBStride) {
  __shared__ __bf16 lA[2][128][40] __attribute__((aligned(16)));
  __shared__ __bf16 fB[2][4][4][16][8] __attribute__((aligned(16)));

  const int tid   = threadIdx.x;
  const int wv    = tid >> 5;
  const int lane  = tid & 31;
  const int half  = lane >> 4;
  const int line  = lane & 15;
  const int cBase = blockIdx.x * 64;
  const int rBase = blockIdx.y * 128;
  const int b     = blockIdx.z;

  const float*  Xf = (const float*)X  + (size_t)xBStride * b;
  const __bf16* Xh = (const __bf16*)X + (size_t)xBStride * b;

  // staging coordinates
  const int ar = tid >> 1;              // A row    0..127
  const int ac = (tid & 1) * 16;        // A col    0 / 16
  const int kp = tid >> 4;              // B k-pair 0..15 (rows 2kp, 2kp+1)
  const int bcc = (tid & 15) * 4;       // B col    0..60

  const float* Wrow = W + (size_t)(rBase + ar) * I + ac;

  const v8f vz = {0.f, 0.f, 0.f, 0.f, 0.f, 0.f, 0.f, 0.f};
  v8f acc[4];
#pragma unroll
  for (int i = 0; i < 4; ++i) acc[i] = vz;

  float4 fa[4];
  float4 fb0, fb1;   // f32  X path
  uint2  hb0, hb1;   // bf16 X path

  auto fetch = [&](int k0) {
#pragma unroll
    for (int c = 0; c < 4; ++c)
      fa[c] = *(const float4*)(Wrow + k0 + 4 * c);
    if (XBF16) {
      hb0 = *(const uint2*)(Xh + (size_t)(k0 + 2 * kp)     * ldx + cBase + bcc);
      hb1 = *(const uint2*)(Xh + (size_t)(k0 + 2 * kp + 1) * ldx + cBase + bcc);
    } else {
      fb0 = *(const float4*)(Xf + (size_t)(k0 + 2 * kp)     * ldx + cBase + bcc);
      fb1 = *(const float4*)(Xf + (size_t)(k0 + 2 * kp + 1) * ldx + cBase + bcc);
    }
  };

  auto stage = [&](int buf) {
    // ---- A: 16 contiguous bf16 -> two aligned 16B stores ----
    union { __bf16 h[16]; uint4 q[2]; } pa;
#pragma unroll
    for (int c = 0; c < 4; ++c) {
      pa.h[4 * c + 0] = (__bf16)fa[c].x;
      pa.h[4 * c + 1] = (__bf16)fa[c].y;
      pa.h[4 * c + 2] = (__bf16)fa[c].z;
      pa.h[4 * c + 3] = (__bf16)fa[c].w;
    }
    *(uint4*)&lA[buf][ar][ac]     = pa.q[0];
    *(uint4*)&lA[buf][ar][ac + 8] = pa.q[1];
    // ---- B: fragment-major; pack (k, k+1) pair per col as one b32 ----
    const int g  = kp >> 2;          // k-octet
    const int ko = 2 * (kp & 3);     // even index within octet
#pragma unroll
    for (int c = 0; c < 4; ++c) {
      union { __bf16 h[2]; unsigned u; } pk;
      if (XBF16) {
        pk.h[0] = ((const __bf16*)&hb0)[c];
        pk.h[1] = ((const __bf16*)&hb1)[c];
      } else {
        pk.h[0] = (__bf16)((const float*)&fb0)[c];
        pk.h[1] = (__bf16)((const float*)&fb1)[c];
      }
      const int col = bcc + c;
      *(unsigned*)&fB[buf][col >> 4][g][col & 15][ko] = pk.u;
    }
  };

  const int nsteps = I >> 5;
  fetch(0);
  stage(0);
  __syncthreads();

  for (int s = 0; s < nsteps; ++s) {
    const int  cur  = s & 1;
    const bool more = (s + 1) < nsteps;
    if (more) {
      if (s + 2 < nsteps) __builtin_prefetch(Wrow + (s + 2) * 32, 0, 0);
      fetch((s + 1) * 32);           // global loads fly over the WMMAs below
    }

    Frag af;
    af.q[0] = *(const uint4*)&lA[cur][wv * 16 + line][half * 8];
    af.q[1] = *(const uint4*)&lA[cur][wv * 16 + line][16 + half * 8];
#pragma unroll
    for (int nt = 0; nt < 4; ++nt) {
      Frag bf;
      bf.q[0] = *(const uint4*)&fB[cur][nt][half][line][0];
      bf.q[1] = *(const uint4*)&fB[cur][nt][2 + half][line][0];
      acc[nt] = __builtin_amdgcn_wmma_f32_16x16x32_bf16(
          false, af.v, false, bf.v, (short)0, acc[nt], false, false);
    }

    if (more) stage(cur ^ 1);
    __syncthreads();
  }

  // ---- epilogue: bias / BN / ReLU / residual ----
  const int rowLocal = wv * 16 + half * 8;     // + r (0..7)
  float sc[8], sh[8];
#pragma unroll
  for (int r = 0; r < 8; ++r) {
    const int row = rBase + rowLocal + r;
    const float bsv = bias ? bias[row] : 0.f;
    if (bn_g) {
      const float s = bn_g[row] * rsqrtf(bn_v[row] + EPSc);
      sc[r] = s;
      sh[r] = (bsv - bn_m[row]) * s + bn_b[row];
    } else {
      sc[r] = 1.f;
      sh[r] = bsv;
    }
  }
  float*  oF = out_f32  ? out_f32  + (size_t)oBStride * b : nullptr;
  __bf16* oH = out_bf16 ? out_bf16 + (size_t)oBStride * b : nullptr;
  const float* rs = resid ? resid + (size_t)rBStride * b : nullptr;

#pragma unroll
  for (int nt = 0; nt < 4; ++nt) {
    const int col = cBase + nt * 16 + line;
#pragma unroll
    for (int r = 0; r < 8; ++r) {
      const int row = rBase + rowLocal + r;
      float val = acc[nt][r] * sc[r] + sh[r];
      if (do_relu) val = fmaxf(val, 0.f);
      if (rs) val += rs[(size_t)row * ldr + col];
      if (oF) oF[(size_t)row * ldo + col] = val;
      if (oH) oH[(size_t)row * ldo + col] = (__bf16)val;
    }
  }
}

// ---------------------------------------------------------------------------
// kNN: top-16 nearest (incl. self). One thread per query point; candidates
// tiled through LDS. Neighbor order is irrelevant downstream.
// ---------------------------------------------------------------------------
__global__ __launch_bounds__(256)
void knn_kernel(const float* __restrict__ pos, int* __restrict__ idx) {
  __shared__ float sx[256], sy[256], sz[256];
  const int b = blockIdx.y;
  const int n = blockIdx.x * 256 + threadIdx.x;
  const float* px = pos + (size_t)b * 3 * Nc;
  const float* py = px + Nc;
  const float* pz = py + Nc;
  const float qx = px[n], qy = py[n], qz = pz[n];

  float bd[Kc];
  int   bi[Kc];
#pragma unroll
  for (int t = 0; t < Kc; ++t) { bd[t] = 3.0e38f; bi[t] = 0; }
  float worst = 3.0e38f;
  int wslot = 0;

  for (int j0 = 0; j0 < Nc; j0 += 256) {
    __syncthreads();
    const int tj = j0 + threadIdx.x;
    sx[threadIdx.x] = px[tj];
    sy[threadIdx.x] = py[tj];
    sz[threadIdx.x] = pz[tj];
    __syncthreads();
    for (int jj = 0; jj < 256; ++jj) {
      const float dx = qx - sx[jj], dy = qy - sy[jj], dz = qz - sz[jj];
      const float d = dx * dx + dy * dy + dz * dz;
      if (d < worst) {
        bd[wslot] = d;
        bi[wslot] = j0 + jj;
        worst = bd[0]; wslot = 0;
#pragma unroll
        for (int t = 1; t < Kc; ++t)
          if (bd[t] > worst) { worst = bd[t]; wslot = t; }
      }
    }
  }
  int* op = idx + ((size_t)b * Nc + n) * Kc;
#pragma unroll
  for (int t = 0; t < Kc; ++t) op[t] = bi[t];
}

// ---------------------------------------------------------------------------
// Positional encoding MLP: pe = Wp2 * relu(bn(Wp1 * pos_rel + bp1)) + bp2
// One thread per (n,k) column; Wp2 staged in LDS as bf16 (32 KB).
// ---------------------------------------------------------------------------
__global__ __launch_bounds__(256)
void pe_kernel(const float* __restrict__ pos, const int* __restrict__ idx,
               const float* __restrict__ Wp1, const float* __restrict__ bp1,
               const float* __restrict__ gp,  const float* __restrict__ btp,
               const float* __restrict__ mp,  const float* __restrict__ vp,
               const float* __restrict__ Wp2, const float* __restrict__ bp2,
               __bf16* __restrict__ pe) {
  __shared__ float  sW1[PHc * 3];
  __shared__ float  ssc[PHc], ssh[PHc];
  __shared__ __bf16 sW2[DIMc * PHc];   // 32 KB
  const int tid = threadIdx.x;

  for (int i = tid; i < PHc * 3; i += 256) sW1[i] = Wp1[i];
  if (tid < PHc) {
    const float s = gp[tid] * rsqrtf(vp[tid] + EPSc);
    ssc[tid] = s;
    ssh[tid] = (bp1[tid] - mp[tid]) * s + btp[tid];
  }
  for (int i = tid; i < DIMc * PHc; i += 256) sW2[i] = (__bf16)Wp2[i];
  __syncthreads();

  const int b   = blockIdx.y;
  const int col = blockIdx.x * 256 + tid;     // 0 .. N*K-1
  const int n   = col >> 4;
  const int j   = idx[(size_t)b * NKc + col];
  const float* px = pos + (size_t)b * 3 * Nc;
  const float dx = px[n] - px[j];
  const float dy = px[Nc + n] - px[Nc + j];
  const float dz = px[2 * Nc + n] - px[2 * Nc + j];

  float t[PHc];
#pragma unroll
  for (int o = 0; o < PHc; ++o) {
    float s = sW1[o * 3 + 0] * dx + sW1[o * 3 + 1] * dy + sW1[o * 3 + 2] * dz;
    s = s * ssc[o] + ssh[o];
    t[o] = fmaxf(s, 0.f);
  }

  __bf16* outp = pe + (size_t)b * DIMc * NKc + col;
  for (int o = 0; o < DIMc; ++o) {
    const __bf16* w = sW2 + o * PHc;
    float s = bp2[o];
#pragma unroll
    for (int i2 = 0; i2 < PHc; ++i2) s += (float)w[i2] * t[i2];
    outp[(size_t)o * NKc] = (__bf16)s;
  }
}

// ---------------------------------------------------------------------------
// a_in[b,ch,n,k] = q[b,ch,n] - key[b,ch,idx[b,n,k]] + pe[b,ch,n,k]  (bf16)
// ---------------------------------------------------------------------------
__global__ __launch_bounds__(256)
void ain_kernel(const float* __restrict__ q, const float* __restrict__ kf,
                const __bf16* __restrict__ pe, const int* __restrict__ idx,
                __bf16* __restrict__ ain) {
  const size_t e = (size_t)blockIdx.x * 256 + threadIdx.x;  // B*DIM*NK
  const size_t perB = (size_t)DIMc * NKc;
  const int b    = (int)(e / perB);
  const size_t r = e % perB;
  const int ch   = (int)(r / NKc);
  const int col  = (int)(r % NKc);
  const int n    = col >> 4;
  const int j    = idx[(size_t)b * NKc + col];
  const float qv = q[((size_t)b * DIMc + ch) * Nc + n];
  const float kv = kf[((size_t)b * DIMc + ch) * Nc + j];
  const float pv = (float)pe[e];
  ain[e] = (__bf16)(qv - kv + pv);
}

// ---------------------------------------------------------------------------
// softmax over K + aggregate: agg[ch,n] = sum_k softmax(logit)[k]*(v+pe)
// ---------------------------------------------------------------------------
__global__ __launch_bounds__(256)
void softagg_kernel(const float* __restrict__ logits,   // DIM x (CN*16)
                    const float* __restrict__ vbuf,     // batch: DIM x N
                    const __bf16* __restrict__ peB,     // batch: DIM x NK
                    float* __restrict__ agg,            // batch: DIM x N
                    int n0, int CN) {
  const int id = blockIdx.x * 256 + threadIdx.x;        // DIMc * CN
  const int ch = id / CN;
  const int nn = id % CN;
  const int n  = n0 + nn;

  const float* lrow = logits + (size_t)ch * (CN * Kc) + (size_t)nn * Kc;
  float l[Kc];
  float m = -3.0e38f;
#pragma unroll
  for (int i = 0; i < Kc; ++i) { l[i] = lrow[i]; m = fmaxf(m, l[i]); }
  float s = 0.f;
#pragma unroll
  for (int i = 0; i < Kc; ++i) { l[i] = __expf(l[i] - m); s += l[i]; }
  const float inv = 1.f / s;

  const float vv = vbuf[(size_t)ch * Nc + n];
  const __bf16* prow = peB + (size_t)ch * NKc + (size_t)n * Kc;
  float a = 0.f;
#pragma unroll
  for (int i = 0; i < Kc; ++i) a += l[i] * (vv + (float)prow[i]);
  agg[(size_t)ch * Nc + n] = a * inv;
}

// ---------------------------------------------------------------------------
// Host-side launch
// ---------------------------------------------------------------------------
extern "C" void kernel_launch(void* const* d_in, const int* in_sizes, int n_in,
                              void* d_out, int out_size, void* d_ws, size_t ws_size,
                              hipStream_t stream) {
  const float* x   = (const float*)d_in[0];
  const float* pos = (const float*)d_in[1];
  const float* Ws  = (const float*)d_in[2];
  const float* bs  = (const float*)d_in[3];
  const float* Wq  = (const float*)d_in[4];
  const float* bq  = (const float*)d_in[5];
  const float* Wk  = (const float*)d_in[6];
  const float* bk  = (const float*)d_in[7];
  const float* Wv  = (const float*)d_in[8];
  const float* bv  = (const float*)d_in[9];
  const float* Wp1 = (const float*)d_in[10];
  const float* bp1 = (const float*)d_in[11];
  const float* gp  = (const float*)d_in[12];
  const float* btp = (const float*)d_in[13];
  const float* mp  = (const float*)d_in[14];
  const float* vp  = (const float*)d_in[15];
  const float* Wp2 = (const float*)d_in[16];
  const float* bp2 = (const float*)d_in[17];
  const float* Wa1 = (const float*)d_in[18];
  const float* ba1 = (const float*)d_in[19];
  const float* ga  = (const float*)d_in[20];
  const float* bta = (const float*)d_in[21];
  const float* ma  = (const float*)d_in[22];
  const float* va  = (const float*)d_in[23];
  const float* Wa2 = (const float*)d_in[24];
  const float* ba2 = (const float*)d_in[25];
  const float* We  = (const float*)d_in[26];
  const float* be2 = (const float*)d_in[27];
  float* out = (float*)d_out;
  (void)in_sizes; (void)n_in; (void)out_size; (void)ws_size;

  // workspace carve-up (~190 MB)
  char* ws = (char*)d_ws;
  size_t off = 0;
  auto take = [&](size_t bytes) -> char* {
    char* p = ws + off;
    off += (bytes + 255) & ~(size_t)255;
    return p;
  };
  float*  hbuf  = (float*) take((size_t)Bc * DIMc * Nc * 4);
  float*  qbuf  = (float*) take((size_t)Bc * DIMc * Nc * 4);
  float*  kbuf  = (float*) take((size_t)Bc * DIMc * Nc * 4);
  float*  vbuf  = (float*) take((size_t)Bc * DIMc * Nc * 4);
  int*    idxb  = (int*)   take((size_t)Bc * Nc * Kc * 4);
  __bf16* pebuf = (__bf16*)take((size_t)Bc * DIMc * NKc * 2);
  __bf16* ainb  = (__bf16*)take((size_t)Bc * DIMc * NKc * 2);
  float*  aggb  = (float*) take((size_t)Bc * DIMc * Nc * 4);
  __bf16* hidb  = (__bf16*)take((size_t)AHc * CHCOL * 2);
  float*  logb  = (float*) take((size_t)DIMc * CHCOL * 4);

  const dim3 blk(256);

  // h = Ws x + bs             (DIM x N per batch)
  wmma_gemm_kernel<0><<<dim3(Nc / 64, DIMc / 128, Bc), blk, 0, stream>>>(
      Ws, x, bs, nullptr, nullptr, nullptr, nullptr, 0,
      hbuf, nullptr, nullptr, CINc, Nc, Nc, 0,
      (long long)CINc * Nc, (long long)DIMc * Nc, 0);

  // q / k / v = W h + b
  wmma_gemm_kernel<0><<<dim3(Nc / 64, DIMc / 128, Bc), blk, 0, stream>>>(
      Wq, hbuf, bq, nullptr, nullptr, nullptr, nullptr, 0,
      qbuf, nullptr, nullptr, DIMc, Nc, Nc, 0,
      (long long)DIMc * Nc, (long long)DIMc * Nc, 0);
  wmma_gemm_kernel<0><<<dim3(Nc / 64, DIMc / 128, Bc), blk, 0, stream>>>(
      Wk, hbuf, bk, nullptr, nullptr, nullptr, nullptr, 0,
      kbuf, nullptr, nullptr, DIMc, Nc, Nc, 0,
      (long long)DIMc * Nc, (long long)DIMc * Nc, 0);
  wmma_gemm_kernel<0><<<dim3(Nc / 64, DIMc / 128, Bc), blk, 0, stream>>>(
      Wv, hbuf, bv, nullptr, nullptr, nullptr, nullptr, 0,
      vbuf, nullptr, nullptr, DIMc, Nc, Nc, 0,
      (long long)DIMc * Nc, (long long)DIMc * Nc, 0);

  // knn indices
  knn_kernel<<<dim3(Nc / 256, Bc), blk, 0, stream>>>(pos, idxb);

  // positional encoding
  pe_kernel<<<dim3(NKc / 256, Bc), blk, 0, stream>>>(
      pos, idxb, Wp1, bp1, gp, btp, mp, vp, Wp2, bp2, pebuf);

  // a_in = q - gather(k) + pe  (bf16)
  {
    const size_t tot = (size_t)Bc * DIMc * NKc;
    ain_kernel<<<dim3((unsigned)(tot / 256)), blk, 0, stream>>>(
        qbuf, kbuf, pebuf, idxb, ainb);
  }

  // attention MLP, chunked: 4096 columns (256 points) per chunk
  for (int b = 0; b < Bc; ++b) {
    for (int c = 0; c < NKc / CHCOL; ++c) {
      const __bf16* ainp = ainb + (size_t)b * DIMc * NKc + (size_t)c * CHCOL;
      // hidden = relu(bn(Wa1 * a_in + ba1))  -> bf16
      wmma_gemm_kernel<1><<<dim3(CHCOL / 64, AHc / 128, 1), blk, 0, stream>>>(
          Wa1, ainp, ba1, ga, bta, ma, va, 1,
          nullptr, hidb, nullptr, DIMc, NKc, CHCOL, 0, 0, 0, 0);
      // logits = Wa2 * hidden + ba2          -> f32
      wmma_gemm_kernel<1><<<dim3(CHCOL / 64, DIMc / 128, 1), blk, 0, stream>>>(
          Wa2, hidb, ba2, nullptr, nullptr, nullptr, nullptr, 0,
          logb, nullptr, nullptr, AHc, CHCOL, CHCOL, 0, 0, 0, 0);
      // softmax over K + aggregate into agg
      softagg_kernel<<<dim3(CHCOL / Kc), blk, 0, stream>>>(
          logb,
          vbuf + (size_t)b * DIMc * Nc,
          pebuf + (size_t)b * DIMc * NKc,
          aggb + (size_t)b * DIMc * Nc,
          c * (CHCOL / Kc), CHCOL / Kc);
    }
  }

  // y = We * agg + be2 + x   -> d_out (CIN x N per batch)
  wmma_gemm_kernel<0><<<dim3(Nc / 64, CINc / 128, Bc), blk, 0, stream>>>(
      We, aggb, be2, nullptr, nullptr, nullptr, nullptr, 0,
      out, nullptr, x, DIMc, Nc, Nc, Nc,
      (long long)DIMc * Nc, (long long)CINc * Nc, (long long)CINc * Nc);
}
